// ConvLSTM_Block_77206332113163
// MI455X (gfx1250) — compile-verified, matrix operand order checked by reading, not run on hardware
//
#include <hip/hip_runtime.h>
#include <hip/hip_bf16.h>
#include <math.h>

// ---------------- problem constants ----------------
#define B_    16
#define T_    16
#define CIN   32
#define HID_  64
#define HH    64
#define WW    64
#define CTOT  96        // CIN + HID
#define OC    256       // 4*HID
#define KBLK  27        // (9 taps * 96 ch) / 32
#define NEG_SLOPE 0.01f

typedef __attribute__((ext_vector_type(16))) __bf16 v16bf;
typedef __attribute__((ext_vector_type(8)))  float  v8f;

#if __has_builtin(__builtin_amdgcn_global_load_async_to_lds_b128) && \
    __has_builtin(__builtin_amdgcn_s_wait_asynccnt)
#define USE_ASYNC_LDS 1
typedef __attribute__((__vector_size__(4 * sizeof(int)))) int v4i_t;
typedef __attribute__((address_space(1))) v4i_t gv4i_t;   // global
typedef __attribute__((address_space(3))) v4i_t lv4i_t;   // LDS
#else
#define USE_ASYNC_LDS 0
#endif

// ---------------- pack x: [B][T][CIN][H][W] f32 -> [T][B][H][W][CIN] bf16 ----
__global__ __launch_bounds__(256) void pack_x_kernel(const float* __restrict__ x,
                                                     __bf16* __restrict__ xp) {
    size_t idx = (size_t)blockIdx.x * 256 + threadIdx.x;   // over T*B*H*W*CIN
    int ci = idx & 31;
    size_t r = idx >> 5;
    int xx = r & 63; r >>= 6;
    int yy = r & 63; r >>= 6;
    int bb = r & 15;
    int tt = (int)(r >> 4);
    size_t src = ((((size_t)bb * T_ + tt) * CIN + ci) * HH + yy) * WW + xx;
    xp[idx] = (__bf16)x[src];
}

// ---------------- pack W into A-fragment order ------------------------------
// Apack[m][kb][lane][e], m=0..15 (oc tile), kb=0..26, lane=0..31, e=0..15
// lane<16 holds M=lane, K elems {0..7,16..23}; lane>=16 holds M=lane-16, K {8..15,24..31}
__global__ __launch_bounds__(256) void pack_w_kernel(const float* __restrict__ W,
                                                     __bf16* __restrict__ Ap) {
    int idx = blockIdx.x * 256 + threadIdx.x;              // over 16*27*32*16
    int e    = idx & 15;
    int lane = (idx >> 4) & 31;
    int kb   = (idx >> 9) % KBLK;
    int m    = idx / (512 * KBLK);
    int half = lane >> 4;
    int K    = e + 8 * half + ((e >= 8) ? 8 : 0);          // 0..31 within block
    int pos  = kb / 3;                                     // tap index 0..8
    int ky   = pos / 3, kx = pos % 3;
    int ci   = (kb % 3) * 32 + K;                          // 0..95
    int oc   = 16 * m + (lane & 15);
    size_t src = (((size_t)oc * CTOT + ci) * 3 + ky) * 3 + kx;
    Ap[idx] = (__bf16)W[src];
}

// ---------------- zero init c (f32) and h0 (bf16) ---------------------------
__global__ __launch_bounds__(256) void init_state_kernel(float* __restrict__ c,
                                                         __bf16* __restrict__ h0) {
    size_t idx = (size_t)blockIdx.x * 256 + threadIdx.x;   // 4,194,304 each
    c[idx]  = 0.0f;
    h0[idx] = (__bf16)0.0f;
}

// ---------------- fused conv (implicit GEMM, WMMA bf16) + LSTM step ---------
// grid.x = B*H = 1024 (one output row per workgroup), 256 threads = 8 waves.
__global__ __launch_bounds__(256) void convlstm_step_kernel(
    const __bf16* __restrict__ xp,     // [B][H][W][CIN]  (this timestep)
    const __bf16* __restrict__ hprev,  // [B][H][W][HID]
    __bf16* __restrict__ hnext,        // [B][H][W][HID]
    float*  __restrict__ cst,          // [B][HID][H][W]
    const __bf16* __restrict__ Ap,     // packed weights
    const float* __restrict__ bias,    // [256]
    float* __restrict__ out,           // [B][HID][H][W]
    int isLast)
{
    __shared__ __align__(16) __bf16 smem[3 * 66 * CTOT];   // 38,016 bytes

    const int tid  = threadIdx.x;
    const int lane = tid & 31;
    const int wave = tid >> 5;
    const int rowId = blockIdx.x;        // 0..1023
    const int bb = rowId >> 6;
    const int y0 = rowId & 63;

    // ---- stage halo rows y0-1..y0+1, cols x=-1..64, 96 channels (bf16) ----
    for (int it = tid; it < 3 * 66 * 12; it += 256) {
        int ch  = it % 12;               // 8-channel chunk (16 bytes)
        int rc  = it / 12;
        int col = rc % 66;
        int row = rc / 66;
        int y = y0 + row - 1;
        int x = col - 1;
        __bf16* lp = smem + ((row * 66 + col) * CTOT + ch * 8);
        const bool inRange = (y >= 0) && (y < HH) && (x >= 0) && (x < WW);
        const int ci0 = ch * 8;
#if USE_ASYNC_LDS
        if (inRange) {
            void* gp = (ci0 < CIN)
                ? (void*)(xp + ((((size_t)bb * HH + y) * WW + x) * CIN + ci0))
                : (void*)(hprev + ((((size_t)bb * HH + y) * WW + x) * HID_ + (ci0 - CIN)));
            __builtin_amdgcn_global_load_async_to_lds_b128((gv4i_t*)gp, (lv4i_t*)lp, 0, 0);
        } else {
            *(uint4*)lp = make_uint4(0u, 0u, 0u, 0u);
        }
#else
        uint4 v = make_uint4(0u, 0u, 0u, 0u);
        if (inRange) {
            if (ci0 < CIN) {
                v = *(const uint4*)(xp + ((((size_t)bb * HH + y) * WW + x) * CIN + ci0));
            } else {
                v = *(const uint4*)(hprev + ((((size_t)bb * HH + y) * WW + x) * HID_ + (ci0 - CIN)));
            }
        }
        *(uint4*)lp = v;
#endif
    }
#if USE_ASYNC_LDS
    __builtin_amdgcn_s_wait_asynccnt(0);
#endif
    __syncthreads();

    // ---- wave work assignment: m-tiles {mw, mw+4, mw+8, mw+12} (one per gate),
    //      n-tiles {nb, nb+1} (16 spatial cols each) ----
    const int mw = wave & 3;
    const int nb = (wave >> 2) * 2;
    const int half = lane >> 4;
    const int colBase = nb * 16 + (lane & 15);   // lane's x for n-tile nb

    v8f acc[4][2];
    for (int g = 0; g < 4; ++g)
        for (int n = 0; n < 2; ++n)
            acc[g][n] = (v8f){0.f, 0.f, 0.f, 0.f, 0.f, 0.f, 0.f, 0.f};

    for (int kb = 0; kb < KBLK; ++kb) {
        const int pos = kb / 3;
        const int ky  = pos / 3;
        const int kx  = pos % 3;
        const int ci0 = (kb % 3) * 32;

        // A fragments: one per gate's m-tile, streamed from packed global
        v16bf afr[4];
        #pragma unroll
        for (int g = 0; g < 4; ++g) {
            const int m = mw + 4 * g;
            afr[g] = *(const v16bf*)(Ap + (((size_t)m * KBLK + kb) * 32 + lane) * 16);
            if (kb + 1 < KBLK)  // uniform branch; global_prefetch_b8 next K-block
                __builtin_prefetch(Ap + (((size_t)m * KBLK + (kb + 1)) * 32 + lane) * 16, 0, 0);
        }

        // B fragments from LDS (two ds_load_b128 each)
        v16bf bfr[2];
        #pragma unroll
        for (int n = 0; n < 2; ++n) {
            const int col = colBase + 16 * n + kx;          // x + kx (halo offset folded)
            const int base = (ky * 66 + col) * CTOT + ci0 + 8 * half;
            union { uint4 q[2]; v16bf v; } u;
            u.q[0] = *(const uint4*)(smem + base);          // K elems e0..7
            u.q[1] = *(const uint4*)(smem + base + 16);     // K elems e8..15
            bfr[n] = u.v;
        }

        #pragma unroll
        for (int g = 0; g < 4; ++g)
            #pragma unroll
            for (int n = 0; n < 2; ++n)
                acc[g][n] = __builtin_amdgcn_wmma_f32_16x16x32_bf16(
                    false, afr[g], false, bfr[n], (short)0, acc[g][n], false, false);
    }

    // ---- epilogue: full LSTM pointwise update in registers ----
    // C/D layout: VGPR r, lane -> M = r + 8*half, N = lane&15
    #pragma unroll
    for (int n = 0; n < 2; ++n) {
        const int x = (nb + n) * 16 + (lane & 15);
        #pragma unroll
        for (int r = 0; r < 8; ++r) {
            const int hid = 16 * mw + r + 8 * half;
            float gi = acc[0][n][r] + bias[hid];
            float gf = acc[1][n][r] + bias[64 + hid];
            float go = acc[2][n][r] + bias[128 + hid];
            float gg = acc[3][n][r] + bias[192 + hid];
            float i_ = 1.0f / (1.0f + __expf(-gi));
            float f_ = 1.0f / (1.0f + __expf(-gf));
            float o_ = 1.0f / (1.0f + __expf(-go));
            float g_ = tanhf(gg);
            size_t cidx = (((size_t)bb * HID_ + hid) * HH + y0) * WW + x;
            float cn = f_ * cst[cidx] + i_ * g_;
            cst[cidx] = cn;
            float hn = o_ * tanhf(cn);
            hnext[(((size_t)bb * HH + y0) * WW + x) * HID_ + hid] = (__bf16)hn;
            if (isLast)
                out[cidx] = (hn >= 0.0f) ? hn : NEG_SLOPE * hn;
        }
    }
}

// ---------------- host-side launcher ----------------------------------------
extern "C" void kernel_launch(void* const* d_in, const int* in_sizes, int n_in,
                              void* d_out, int out_size, void* d_ws, size_t ws_size,
                              hipStream_t stream) {
    const float* x = (const float*)d_in[0];   // [B][T][CIN][H][W]
    const float* W = (const float*)d_in[1];   // [256][96][3][3]
    const float* b = (const float*)d_in[2];   // [256]
    float* out = (float*)d_out;               // [B][HID][H][W]

    // workspace carve-up (all offsets 256B-aligned)
    char* ws = (char*)d_ws;
    __bf16* xpack = (__bf16*)(ws);                         // 67,108,864 B
    __bf16* Apack = (__bf16*)(ws + 67108864);              //    442,368 B
    __bf16* hA    = (__bf16*)(ws + 67551232);              //  8,388,608 B
    __bf16* hB    = (__bf16*)(ws + 75939840);              //  8,388,608 B
    float*  cst   = (float*) (ws + 84328448);              // 16,777,216 B

    const size_t xElems = (size_t)T_ * B_ * HH * WW * CIN; // 33,554,432
    pack_x_kernel<<<(unsigned)(xElems / 256), 256, 0, stream>>>(x, xpack);

    const int wElems = 16 * KBLK * 32 * 16;                // 221,184
    pack_w_kernel<<<wElems / 256, 256, 0, stream>>>(W, Apack);

    const size_t sElems = (size_t)B_ * HID_ * HH * WW;     // 4,194,304
    init_state_kernel<<<(unsigned)(sElems / 256), 256, 0, stream>>>(cst, hA);

    __bf16* hp = hA;
    __bf16* hn = hB;
    const size_t xStepElems = (size_t)B_ * HH * WW * CIN;  // 2,097,152
    for (int t = 0; t < T_; ++t) {
        convlstm_step_kernel<<<B_ * HH, 256, 0, stream>>>(
            xpack + (size_t)t * xStepElems, hp, hn, cst, Apack, b, out,
            (t == T_ - 1) ? 1 : 0);
        __bf16* tmp = hp; hp = hn; hn = tmp;
    }
}
